// BidirLSTMLayer_59665685676620
// MI455X (gfx1250) — matrix operand order repaired and checked by reading, not compiled, not run on hardware
//
#include <hip/hip_runtime.h>
#include <hip/hip_bf16.h>

// ---------------------------------------------------------------------------
// Bidirectional LSTM, T=1024, B=64, I=H=512, persistent-kernel formulation.
//
// gates = [h, x_t] @ [w_hh ; w_ih]^T + (b_ih + b_hh)     (K = 1024)
//
//  * 32 persistent WGs: 16/direction, each owns 32 h-columns (128 gate rows).
//  * Weights pre-packed to bf16 in ws in the exact padded LDS image layout,
//    then DMA'd into 258 KB of LDS by one TENSOR_LOAD_TO_LDS per WG (TDM,
//    TENSORcnt) -- the 1024-step loop never touches global for weights.
//  * bf16 WMMA 16x16x32, fp32 accumulate; c lives in VGPRs for the whole
//    scan; gate math is lane-aligned in the C/D fragment layout.
//  * One agent-scope barrier per step per direction (monotonic counter in
//    d_ws, zeroed each launch -> graph-safe); h ping-pongs as bf16 in ws.
//  * All hot-loop global accesses forced to address_space(1) so A-fragments
//    are global_load_b128 (not flat_load) and don't touch the LDS path.
// ---------------------------------------------------------------------------

typedef __bf16 bf16;
typedef __attribute__((ext_vector_type(16))) __bf16 v16bf;
typedef __attribute__((ext_vector_type(8)))  float  v8f;
typedef unsigned int v4u   __attribute__((ext_vector_type(4)));
typedef int          v4i   __attribute__((ext_vector_type(4)));
typedef int          v8i   __attribute__((ext_vector_type(8)));
typedef unsigned int u32x4 __attribute__((ext_vector_type(4)));  // POD 16B
typedef unsigned int u32x2 __attribute__((ext_vector_type(2)));  // POD 8B

#define T_STEPS   1024
#define BATCH     64
#define INP       512
#define HID       512
#define KDIM      (HID + INP)          // 1024
#define NWG_DIR   16                   // workgroups per direction
#define COLS_WG   (HID / NWG_DIR)      // 32 h-columns per WG
#define GROWS_WG  (4 * COLS_WG)        // 128 gate rows per WG
#define LDS_PAD   8                    // bf16 elements of row padding
#define LDS_STRIDE (KDIM + LDS_PAD)    // 1032 bf16 per row (516 dwords: no
                                       // LDS bank conflicts on B-frag reads)
#define WBLK_ELEMS (GROWS_WG * LDS_STRIDE)      // 132,096 bf16 per WG block
#define LDS_BYTES  (WBLK_ELEMS * 2)             // 264,192 B (< 320 KB/WGP)

// ---- forced-global accessors (avoid flat_load/flat_store in the hot loop)
__device__ __forceinline__ u32x4 gload16(const void* p) {
    return *(const __attribute__((address_space(1))) u32x4*)p;
}
__device__ __forceinline__ void gstore_f32(void* p, float v) {
    *(__attribute__((address_space(1))) float*)p = v;
}
__device__ __forceinline__ void gstore_bf16(void* p, bf16 v) {
    *(__attribute__((address_space(1))) bf16*)p = v;
}

__device__ __forceinline__ float sig_f(float x) {
    return 1.0f / (1.0f + __expf(-x));
}
__device__ __forceinline__ float tanh_f(float x) {
    return 2.0f / (1.0f + __expf(-2.0f * x)) - 1.0f;
}

// agent-scope arrive + spin-wait on a monotonic counter
__device__ __forceinline__ void grid_barrier(unsigned* bar, unsigned target) {
    __threadfence();
    __syncthreads();
    if (threadIdx.x == 0) {
        __hip_atomic_fetch_add(bar, 1u, __ATOMIC_RELEASE, __HIP_MEMORY_SCOPE_AGENT);
        while (__hip_atomic_load(bar, __ATOMIC_ACQUIRE, __HIP_MEMORY_SCOPE_AGENT) < target)
            __builtin_amdgcn_s_sleep(2);
    }
    __syncthreads();
}

// ---------------------------------------------------------------------------
__global__ void init_ws_kernel(unsigned* bar) {
    if (threadIdx.x < 8) bar[threadIdx.x] = 0u;
}

// fp32 -> bf16 bulk convert of x, float4 -> packed 4xbf16
__global__ void cvt_x_kernel(const float* __restrict__ x,
                             bf16* __restrict__ xb, int n4) {
    int i      = blockIdx.x * blockDim.x + threadIdx.x;
    int stride = gridDim.x * blockDim.x;
    for (; i < n4; i += stride) {
        float4 f = ((const float4*)x)[i];
        union { bf16 h[4]; u32x2 u; } p;
        p.h[0] = (bf16)f.x; p.h[1] = (bf16)f.y;
        p.h[2] = (bf16)f.z; p.h[3] = (bf16)f.w;
        ((u32x2*)xb)[i] = p.u;
    }
}

// Pre-pack W_cat = [w_hh | w_ih] (fp32 -> bf16) into ws as the *exact LDS
// image* each WG will DMA: per-WG-contiguous 128 rows x LDS_STRIDE elements.
__global__ void cvt_w_kernel(const float* __restrict__ w_ih_f,
                             const float* __restrict__ w_hh_f,
                             const float* __restrict__ w_ih_b,
                             const float* __restrict__ w_hh_b,
                             bf16* __restrict__ wpack) {
    const int wg  = blockIdx.x;      // 0..31 == (dir, cb)
    const int dir = wg >> 4;
    const int cb  = wg & 15;
    const float* w_ih = dir ? w_ih_b : w_ih_f;
    const float* w_hh = dir ? w_hh_b : w_hh_f;
    bf16* dst = wpack + (size_t)wg * WBLK_ELEMS;

    const int kchunks = KDIM / 8;                  // 128 chunks of 8 per row
    for (int chk = threadIdx.x; chk < GROWS_WG * kchunks; chk += blockDim.x) {
        int r    = chk / kchunks;
        int kk   = (chk % kchunks) * 8;
        int gate = r / COLS_WG;
        int c    = r % COLS_WG;
        int grow = gate * HID + cb * COLS_WG + c;  // global gate row
        const float* src = (kk < HID)
            ? (w_hh + (size_t)grow * HID + kk)
            : (w_ih + (size_t)grow * INP + (kk - HID));
        float4 f0 = ((const float4*)src)[0];
        float4 f1 = ((const float4*)src)[1];
        union { bf16 h[8]; u32x4 u; } p;
        p.h[0] = (bf16)f0.x; p.h[1] = (bf16)f0.y;
        p.h[2] = (bf16)f0.z; p.h[3] = (bf16)f0.w;
        p.h[4] = (bf16)f1.x; p.h[5] = (bf16)f1.y;
        p.h[6] = (bf16)f1.z; p.h[7] = (bf16)f1.w;
        *(u32x4*)&dst[(size_t)r * LDS_STRIDE + kk] = p.u;
    }
}

// ---------------------------------------------------------------------------
// TDM: DMA this WG's pre-packed 258 KB weight image into LDS as one 1-D
// tensor of 33,024 x 8-byte elements.  Falls back to a plain copy if the
// builtin is unavailable.
__device__ __forceinline__ void stage_weights_tdm(const bf16* gsrc,
                                                  bf16* lds, int tid) {
#if __has_builtin(__builtin_amdgcn_tensor_load_to_lds)
    if ((tid >> 5) == 0) {                        // one wave issues the DMA
        const unsigned long long ga = (unsigned long long)gsrc;
        const unsigned nqw = (unsigned)(LDS_BYTES / 8);   // 33,024 qwords

        v4u g0;                                    // D# group 0
        g0.x = 1u;                                 // count=1, user mode
        g0.y = 0u;                                 // lds_addr = 0 (our base)
        g0.z = (unsigned)(ga & 0xFFFFFFFFu);       // global_addr[31:0]
        g0.w = ((unsigned)(ga >> 32) & 0x01FFFFFFu) | (2u << 30); // type=2

        v8i g1;                                    // D# group 1
        g1[0] = (int)(3u << 16);                   // wg_mask=0, data_size=8B
        g1[1] = (int)((nqw & 0xFFFFu) << 16);      // tensor_dim0[15:0]
        g1[2] = (int)(((nqw >> 16) & 0xFFFFu) | (1u << 16)); // dim0 hi, dim1=1
        g1[3] = (int)((nqw & 0xFFFFu) << 16);      // tile_dim0 = nqw
        g1[4] = 0;                                 // tile_dim1/2 unused (1-D)
        g1[5] = (int)nqw;                          // tensor_dim0_stride lo
        g1[6] = 0;
        g1[7] = 0;

        v4i gz = {0, 0, 0, 0};
#if defined(__clang_major__) && (__clang_major__ >= 23)
        v8i gz8 = {0, 0, 0, 0, 0, 0, 0, 0};
        __builtin_amdgcn_tensor_load_to_lds(g0, g1, gz, gz, gz8, 0);
#else
        __builtin_amdgcn_tensor_load_to_lds(g0, g1, gz, gz, 0);
#endif
#if __has_builtin(__builtin_amdgcn_s_wait_tensorcnt)
        __builtin_amdgcn_s_wait_tensorcnt(0);
#endif
    }
    __syncthreads();
#else
    // fallback: cooperative vector copy
    const u32x4* s = (const u32x4*)gsrc;
    u32x4*       d = (u32x4*)lds;
    for (int i = tid; i < LDS_BYTES / 16; i += (int)blockDim.x) d[i] = s[i];
    __syncthreads();
#endif
}

// ---------------------------------------------------------------------------
__global__ __launch_bounds__(256, 1)
void lstm_persistent_kernel(const bf16*  __restrict__ x_bf,
                            const bf16*  __restrict__ wpack,
                            const float* __restrict__ h0_f, const float* __restrict__ c0_f,
                            const float* __restrict__ h0_b, const float* __restrict__ c0_b,
                            const float* __restrict__ b_ih_f, const float* __restrict__ b_hh_f,
                            const float* __restrict__ b_ih_b, const float* __restrict__ b_hh_b,
                            bf16*  __restrict__ hbuf,   // [2 dir][2 buf][B*H] bf16
                            unsigned* __restrict__ bar, // [>=2] counters
                            float* __restrict__ out)    // T*B*2H ++ hf,cf,hb,cb
{
    const int wg   = blockIdx.x;     // 0..31
    const int dir  = wg >> 4;        // 0 = forward, 1 = backward
    const int cb   = wg & 15;        // h-column block (32 cols)
    const int tid  = threadIdx.x;
    const int wave = tid >> 5;
    const int lane = tid & 31;

    const float* b_ih = dir ? b_ih_b : b_ih_f;
    const float* b_hh = dir ? b_hh_b : b_hh_f;
    const float* h0   = dir ? h0_b   : h0_f;
    const float* c0   = dir ? c0_b   : c0_f;
    unsigned*   mybar = bar + dir;

    bf16* hb[2];
    hb[0] = hbuf + (size_t)dir * 2 * BATCH * HID;
    hb[1] = hb[0] + BATCH * HID;

    extern __shared__ bf16 wlds[];   // [GROWS_WG][LDS_STRIDE]

    // ---- one TDM DMA pulls the pre-packed weight image into LDS
    stage_weights_tdm(wpack + (size_t)wg * WBLK_ELEMS, wlds, tid);

    // ---- per-wave tile assignment
    const int rt   = wave & 3;       // batch row tile: rows rt*16 .. rt*16+15
    const int chh  = wave >> 2;      // column half within the 32-col block
    const int nloc = lane & 15;      // N inside 16x16 tile
    const int lhi  = lane >> 4;      // lane-half (A/B/C fragment layouts)
    const int colg = cb * COLS_WG + chh * 16 + nloc;   // global h column

    float bias[4];
#pragma unroll
    for (int g = 0; g < 4; ++g)
        bias[g] = b_ih[g * HID + colg] + b_hh[g * HID + colg];

    // ---- stage h0 (fp32 -> bf16) for this WG's column slice
    for (int i = tid; i < BATCH * COLS_WG; i += blockDim.x) {
        int b = i / COLS_WG, c = i % COLS_WG;
        int col = cb * COLS_WG + c;
        hb[0][b * HID + col] = (bf16)h0[b * HID + col];
    }

    // ---- c state lives in registers for the entire scan
    // C/D layout: VGPR v, lanes 0-15 -> M=v, lanes 16-31 -> M=v+8, N=lane&15
    v8f creg;
#pragma unroll
    for (int v = 0; v < 8; ++v) {
        int b = rt * 16 + v + lhi * 8;
        creg[v] = c0[b * HID + colg];
    }

    grid_barrier(mybar, NWG_DIR * 1);   // h0 staged in all 16 peer WGs

    const int arow = rt * 16 + (lane & 15);   // A-matrix row for this lane
    // B-fragment LDS base for each gate (lane N = lane&15 holds a W_cat row)
    const bf16* bbase[4];
#pragma unroll
    for (int g = 0; g < 4; ++g)
        bbase[g] = &wlds[(size_t)(g * COLS_WG + chh * 16 + nloc) * LDS_STRIDE
                         + lhi * 8];

    // ======================= serial time scan ============================
    for (int s = 0; s < T_STEPS; ++s) {
        const int t = dir ? (T_STEPS - 1 - s) : s;
        const bf16* hcur  = hb[s & 1];
        bf16*       hnext = hb[(s + 1) & 1];
        const bf16* xt    = x_bf + (size_t)t * BATCH * INP;

        const bf16* hrowp = hcur + (size_t)arow * HID + lhi * 8;
        const bf16* xrowp = xt   + (size_t)arow * INP + lhi * 8;

        v8f acc[4];
#pragma unroll
        for (int g = 0; g < 4; ++g)
#pragma unroll
            for (int v = 0; v < 8; ++v) acc[g][v] = bias[g];

        // ---- K-half 1: A = h_{t-1}   (global_load_b128 from L2)
#pragma unroll 2
        for (int ks = 0; ks < 16; ++ks) {
            const int k0 = ks * 32;
            v16bf afrag;
            ((u32x4*)&afrag)[0] = gload16(hrowp + k0);
            ((u32x4*)&afrag)[1] = gload16(hrowp + k0 + 16);
#pragma unroll
            for (int g = 0; g < 4; ++g) {
                const bf16* bsrc = bbase[g] + k0;
                v16bf bfrag;
                ((u32x4*)&bfrag)[0] = ((const u32x4*)bsrc)[0];
                ((u32x4*)&bfrag)[1] = ((const u32x4*)bsrc)[2];
                acc[g] = __builtin_amdgcn_wmma_f32_16x16x32_bf16(
                    false, afrag, false, bfrag, (short)0, acc[g], false, false);
            }
        }
        // ---- K-half 2: A = x_t
#pragma unroll 2
        for (int ks = 0; ks < 16; ++ks) {
            const int k0 = ks * 32;
            v16bf afrag;
            ((u32x4*)&afrag)[0] = gload16(xrowp + k0);
            ((u32x4*)&afrag)[1] = gload16(xrowp + k0 + 16);
#pragma unroll
            for (int g = 0; g < 4; ++g) {
                const bf16* bsrc = bbase[g] + HID + k0;
                v16bf bfrag;
                ((u32x4*)&bfrag)[0] = ((const u32x4*)bsrc)[0];
                ((u32x4*)&bfrag)[1] = ((const u32x4*)bsrc)[2];
                acc[g] = __builtin_amdgcn_wmma_f32_16x16x32_bf16(
                    false, afrag, false, bfrag, (short)0, acc[g], false, false);
            }
        }

        // ---- gate nonlinearities + cell update (lane-aligned)
        float* outp = out + (size_t)t * BATCH * 2 * HID;
#pragma unroll
        for (int v = 0; v < 8; ++v) {
            float ig = sig_f(acc[0][v]);
            float fg = sig_f(acc[1][v]);
            float gg = tanh_f(acc[2][v]);
            float og = sig_f(acc[3][v]);
            float cn = fg * creg[v] + ig * gg;
            creg[v]  = cn;
            float hn = og * tanh_f(cn);
            int b = rt * 16 + v + lhi * 8;
            gstore_f32(&outp[(size_t)b * 2 * HID + dir * HID + colg], hn);
            gstore_bf16(&hnext[b * HID + colg], (bf16)hn);
        }

        // prefetch next timestep's x row (gfx1250 global_prefetch_b8)
        if (s + 1 < T_STEPS) {
            const int tn = dir ? (T_STEPS - 2 - s) : (s + 1);
            __builtin_prefetch(x_bf + (size_t)tn * BATCH * INP
                                    + (size_t)arow * INP, 0, 1);
        }

        grid_barrier(mybar, NWG_DIR * (s + 2));
    }

    // ---- final h_T, c_T outputs (order: hf, cf, hb, cb after the sequence)
    float* hout = out + (size_t)T_STEPS * BATCH * 2 * HID
                      + (size_t)dir * 2 * BATCH * HID;
    float* cout = hout + BATCH * HID;
    const int t_fin = dir ? 0 : (T_STEPS - 1);
#pragma unroll
    for (int v = 0; v < 8; ++v) {
        int b = rt * 16 + v + lhi * 8;
        hout[b * HID + colg] =
            out[(size_t)t_fin * BATCH * 2 * HID + (size_t)b * 2 * HID
                + dir * HID + colg];
        cout[b * HID + colg] = creg[v];
    }
}

// ---------------------------------------------------------------------------
extern "C" void kernel_launch(void* const* d_in, const int* in_sizes, int n_in,
                              void* d_out, int out_size, void* d_ws, size_t ws_size,
                              hipStream_t stream) {
    const float* x      = (const float*)d_in[0];
    const float* h0_f   = (const float*)d_in[1];
    const float* c0_f   = (const float*)d_in[2];
    const float* h0_b   = (const float*)d_in[3];
    const float* c0_b   = (const float*)d_in[4];
    const float* w_ih_f = (const float*)d_in[5];
    const float* w_hh_f = (const float*)d_in[6];
    const float* b_ih_f = (const float*)d_in[7];
    const float* b_hh_f = (const float*)d_in[8];
    const float* w_ih_b = (const float*)d_in[9];
    const float* w_hh_b = (const float*)d_in[10];
    const float* b_ih_b = (const float*)d_in[11];
    const float* b_hh_b = (const float*)d_in[12];
    float* out = (float*)d_out;

    // workspace layout
    char*     ws    = (char*)d_ws;
    unsigned* bar   = (unsigned*)ws;                               // 256 B
    bf16*     hbuf  = (bf16*)(ws + 256);                           // 256 KB
    bf16*     x_bf  = (bf16*)(ws + 256 + 2 * 2 * BATCH * HID * sizeof(bf16));
    bf16*     wpack = x_bf + (size_t)T_STEPS * BATCH * INP;        // 8.25 MB

    init_ws_kernel<<<1, 64, 0, stream>>>(bar);

    const int n4 = T_STEPS * BATCH * INP / 4;
    cvt_x_kernel<<<2048, 256, 0, stream>>>(x, x_bf, n4);
    cvt_w_kernel<<<dim3(2 * NWG_DIR), dim3(256), 0, stream>>>(
        w_ih_f, w_hh_f, w_ih_b, w_hh_b, wpack);

    lstm_persistent_kernel<<<dim3(2 * NWG_DIR), dim3(256), LDS_BYTES, stream>>>(
        x_bf, wpack, h0_f, c0_f, h0_b, c0_b,
        b_ih_f, b_hh_f, b_ih_b, b_hh_b,
        hbuf, bar, out);
}
